// MoELayer_4767413699146
// MI455X (gfx1250) — matrix operand order repaired and checked by reading, not compiled, not run on hardware
//
#include <hip/hip_runtime.h>
#include <hip/hip_bf16.h>
#include <stdint.h>

#define TOKENS 4096   // B*S
#define DIM    1024   // D
#define NEXP   8      // E
#define FFN    4096   // F
#define MBLK   128    // M rows per workgroup
#define MAXMB  (TOKENS / MBLK)   // 32 worst-case M-blocks per expert

typedef __attribute__((ext_vector_type(16))) __bf16   v16bf;
typedef __attribute__((ext_vector_type(2)))  __bf16   v2bf;
typedef __attribute__((ext_vector_type(8)))  float    v8f;
typedef __attribute__((ext_vector_type(8)))  uint32_t v8u;

union FragU { v8u u; v16bf b; };

__device__ __forceinline__ uint32_t f2bf(float f) {
    uint32_t u = __float_as_uint(f);
    return (u + 0x7FFFu + ((u >> 16) & 1u)) >> 16;   // RNE
}
__device__ __forceinline__ uint32_t pack2(float lo, float hi) {
#if __has_builtin(__builtin_amdgcn_cvt_pk_bf16_f32)
    v2bf r = __builtin_amdgcn_cvt_pk_bf16_f32(lo, hi);
    return __builtin_bit_cast(uint32_t, r);
#else
    return f2bf(lo) | (f2bf(hi) << 16);
#endif
}
__device__ __forceinline__ float gelu_tanh(float x) {
    float x3 = x * x * x;
    return 0.5f * x * (1.0f + tanhf(0.7978845608028654f * (x + 0.044715f * x3)));
}

#if __has_builtin(__builtin_amdgcn_global_load_async_to_lds_b32)
#define HAVE_ASYNC_LDS 1
typedef __attribute__((address_space(1))) int* as1i;   // prints as '__device__ int *'
typedef __attribute__((address_space(3))) int* as3i;   // prints as '__shared__ int *'
#endif

__device__ __forceinline__ void wait_async_zero() {
#if defined(HAVE_ASYNC_LDS)
#if __has_builtin(__builtin_amdgcn_s_wait_asynccnt)
    __builtin_amdgcn_s_wait_asynccnt(0);
#else
    asm volatile("s_wait_asynccnt 0" ::: "memory");
#endif
#endif
}

// ---------------------------------------------------------------- init
__global__ void moe_init(int* cnt, float* psum, int* cursor) {
    int i = threadIdx.x;
    if (i < NEXP) { cnt[i] = 0; psum[i] = 0.0f; cursor[i] = 0; }
}

// ---------------------------------------------------------------- gate: logits, softmax, argmax
// one wave (32 lanes) per token; 8 waves per block; LDS-aggregated atomics
__global__ __launch_bounds__(256) void moe_gate(const float* __restrict__ X,
                                                const float* __restrict__ Wg,
                                                int* __restrict__ eid,
                                                float* __restrict__ pg,
                                                int* __restrict__ cnt,
                                                float* __restrict__ psum) {
    __shared__ float s_psum[NEXP];
    __shared__ int   s_cnt[NEXP];
    const int lane = threadIdx.x & 31;
    const int wid  = threadIdx.x >> 5;
    const int t    = blockIdx.x * 8 + wid;

    if (threadIdx.x < NEXP) { s_psum[threadIdx.x] = 0.0f; s_cnt[threadIdx.x] = 0; }
    __syncthreads();

    float acc[NEXP];
#pragma unroll
    for (int e = 0; e < NEXP; ++e) acc[e] = 0.0f;

    const float* xrow = X + (size_t)t * DIM;
    for (int d = lane; d < DIM; d += 32) {
        float xv = xrow[d];
        const float* wrow = Wg + d * NEXP;
#pragma unroll
        for (int e = 0; e < NEXP; ++e) acc[e] += xv * wrow[e];
    }
#pragma unroll
    for (int off = 16; off > 0; off >>= 1) {
#pragma unroll
        for (int e = 0; e < NEXP; ++e) acc[e] += __shfl_xor(acc[e], off, 32);
    }
    if (lane == 0) {
        float m = acc[0]; int g = 0;
#pragma unroll
        for (int e = 1; e < NEXP; ++e) if (acc[e] > m) { m = acc[e]; g = e; }
        float ex[NEXP], s = 0.0f;
#pragma unroll
        for (int e = 0; e < NEXP; ++e) { ex[e] = expf(acc[e] - m); s += ex[e]; }
        float inv = 1.0f / s;
#pragma unroll
        for (int e = 0; e < NEXP; ++e) atomicAdd(&s_psum[e], ex[e] * inv);
        pg[t]  = ex[g] * inv;
        eid[t] = g;
        atomicAdd(&s_cnt[g], 1);
    }
    __syncthreads();
    if (threadIdx.x < NEXP) {
        atomicAdd(psum + threadIdx.x, s_psum[threadIdx.x]);
        atomicAdd(cnt + threadIdx.x, s_cnt[threadIdx.x]);
    }
}

// ---------------------------------------------------------------- scan + balance loss + gate_load
__global__ void moe_scan(const int* __restrict__ cnt, const float* __restrict__ psum,
                         int* __restrict__ offs, float* __restrict__ out) {
    if (threadIdx.x == 0) {
        int tot = 0;
        float loss = 0.0f;
#pragma unroll
        for (int e = 0; e < NEXP; ++e) { offs[e] = tot; tot += cnt[e]; }
#pragma unroll
        for (int e = 0; e < NEXP; ++e)
            loss += (psum[e] / (float)TOKENS) * ((float)cnt[e] / (float)TOKENS);
        loss *= (float)NEXP;
        out[(size_t)TOKENS * DIM] = loss;
        int* gl = (int*)(out + (size_t)TOKENS * DIM + 1);
#pragma unroll
        for (int e = 0; e < NEXP; ++e) gl[e] = cnt[e];
    }
}

// ---------------------------------------------------------------- scatter token ids per expert
__global__ __launch_bounds__(256) void moe_scatter(const int* __restrict__ eid,
                                                   int* __restrict__ cursor,
                                                   const int* __restrict__ offs,
                                                   int* __restrict__ tl) {
    int t = blockIdx.x * 256 + threadIdx.x;
    int e = eid[t];
    int p = atomicAdd(cursor + e, 1);
    tl[offs[e] + p] = t;
}

// ---------------------------------------------------------------- GEMM1: h = gelu(x @ W1[e] + b1[e])  (bf16 out)
// block: 256 thr = 8 waves; tile 128(M) x 128(N); K-step 32
__global__ __launch_bounds__(256) void moe_ffn1(const float* __restrict__ X,
                                                const float* __restrict__ W1,
                                                const float* __restrict__ b1,
                                                const int* __restrict__ cnt,
                                                const int* __restrict__ offs,
                                                const int* __restrict__ tl,
                                                unsigned short* __restrict__ h) {
    __shared__ uint32_t As[MBLK * 16];    // 128 rows x 16 dwords (32 bf16 K)
    __shared__ uint32_t Bs[128 * 16];     // 128 cols x 16 dwords (32 bf16 K)
    __shared__ int toks[MBLK];

    const int e  = blockIdx.x >> 5;       // /MAXMB
    const int mb = blockIdx.x & (MAXMB - 1);
    const int cnt_e = cnt[e];
    if (mb * MBLK >= cnt_e) return;
    const int off_e = offs[e];
    const int nb  = blockIdx.y;           // N tile of 128 (F/128 = 32)
    const int tid = threadIdx.x;
    const int lane = tid & 31, wm = tid >> 5;   // wave id = M stripe
    const int hl = lane >> 4, l16 = lane & 15;

    if (tid < MBLK) {
        int r = mb * MBLK + tid;
        toks[tid] = tl[off_e + min(r, cnt_e - 1)];   // clamp: branchless staging
    }
    __syncthreads();

    v8f acc[8];
#pragma unroll
    for (int ni = 0; ni < 8; ++ni)
#pragma unroll
        for (int j = 0; j < 8; ++j) acc[ni][j] = 0.0f;

    const float* Bsrc = W1 + (size_t)e * DIM * FFN + (size_t)nb * 128;

    for (int k0 = 0; k0 < DIM; k0 += 32) {
        // stage A (gather rows, f32 -> bf16): 2048 dwords
#pragma unroll
        for (int i = 0; i < 8; ++i) {
            int idx = tid + i * 256;
            int row = idx >> 4, kd = idx & 15;
            int t = toks[row];
            const float2 p = *(const float2*)(X + (size_t)t * DIM + k0 + kd * 2);
            As[idx] = pack2(p.x, p.y);
        }
        // stage B (f32 -> bf16, K pairs packed per dword): 2048 dwords
#pragma unroll
        for (int i = 0; i < 8; ++i) {
            int idx = tid + i * 256;
            int kd = idx >> 7, n = idx & 127;
            const float* bp = Bsrc + (size_t)(k0 + kd * 2) * FFN + n;
            if (k0 + 32 < DIM) __builtin_prefetch(bp + (size_t)32 * FFN, 0, 0);
            Bs[n * 16 + kd] = pack2(bp[0], bp[FFN]);
        }
        __syncthreads();

        FragU fa;
        const int arow = wm * 16 + l16;
#pragma unroll
        for (int j = 0; j < 4; ++j) fa.u[j]     = As[arow * 16 + hl * 4 + j];
#pragma unroll
        for (int j = 0; j < 4; ++j) fa.u[4 + j] = As[arow * 16 + 8 + hl * 4 + j];

#pragma unroll
        for (int ni = 0; ni < 8; ++ni) {
            FragU fb;
            const int bcol = ni * 16 + l16;
#pragma unroll
            for (int j = 0; j < 8; ++j) fb.u[j] = Bs[bcol * 16 + (j >> 2) * 8 + hl * 4 + (j & 3)];
            acc[ni] = __builtin_amdgcn_wmma_f32_16x16x32_bf16(
                false, fa.b, false, fb.b, (short)0, acc[ni], false, false);
        }
        __syncthreads();
    }

#pragma unroll
    for (int ni = 0; ni < 8; ++ni) {
        const int f = nb * 128 + ni * 16 + l16;
        const float bias = b1[e * FFN + f];
#pragma unroll
        for (int j = 0; j < 8; ++j) {
            int rl = wm * 16 + hl * 8 + j;
            if (mb * MBLK + rl < cnt_e) {
                int t = toks[rl];
                float v = gelu_tanh(acc[ni][j] + bias);
                h[(size_t)t * FFN + f] = (unsigned short)f2bf(v);
            }
        }
    }
}

// ---------------------------------------------------------------- GEMM2: out = pg * (h @ W2[e] + b2[e])
__global__ __launch_bounds__(256) void moe_ffn2(const uint32_t* __restrict__ h32,
                                                const float* __restrict__ W2,
                                                const float* __restrict__ b2,
                                                const int* __restrict__ cnt,
                                                const int* __restrict__ offs,
                                                const int* __restrict__ tl,
                                                const float* __restrict__ pg,
                                                float* __restrict__ out) {
    __shared__ uint32_t As[MBLK * 16];
    __shared__ uint32_t Bs[128 * 16];
    __shared__ int toks[MBLK];

    const int e  = blockIdx.x >> 5;
    const int mb = blockIdx.x & (MAXMB - 1);
    const int cnt_e = cnt[e];
    if (mb * MBLK >= cnt_e) return;
    const int off_e = offs[e];
    const int nb  = blockIdx.y;           // N tile of 128 (D/128 = 8)
    const int tid = threadIdx.x;
    const int lane = tid & 31, wm = tid >> 5;
    const int hl = lane >> 4, l16 = lane & 15;

    if (tid < MBLK) {
        int r = mb * MBLK + tid;
        toks[tid] = tl[off_e + min(r, cnt_e - 1)];
    }
    __syncthreads();

    v8f acc[8];
#pragma unroll
    for (int ni = 0; ni < 8; ++ni)
#pragma unroll
        for (int j = 0; j < 8; ++j) acc[ni][j] = 0.0f;

    const float* Bsrc = W2 + (size_t)e * FFN * DIM + (size_t)nb * 128;

    for (int k0 = 0; k0 < FFN; k0 += 32) {
        // stage A directly from bf16 h (pure byte copy -> async global->LDS if available)
#pragma unroll
        for (int i = 0; i < 8; ++i) {
            int idx = tid + i * 256;
            int row = idx >> 4, kd = idx & 15;
            int t = toks[row];
            const uint32_t* gp = h32 + (size_t)t * (FFN / 2) + (k0 >> 1) + kd;
#if defined(HAVE_ASYNC_LDS)
            __builtin_amdgcn_global_load_async_to_lds_b32(
                (as1i)(int*)gp, (as3i)(int*)&As[idx], 0, 0);
#else
            As[idx] = *gp;
#endif
        }
        // stage B (f32 -> bf16)
#pragma unroll
        for (int i = 0; i < 8; ++i) {
            int idx = tid + i * 256;
            int kd = idx >> 7, n = idx & 127;
            const float* bp = Bsrc + (size_t)(k0 + kd * 2) * DIM + n;
            if (k0 + 32 < FFN) __builtin_prefetch(bp + (size_t)32 * DIM, 0, 0);
            Bs[n * 16 + kd] = pack2(bp[0], bp[DIM]);
        }
        wait_async_zero();
        __syncthreads();

        FragU fa;
        const int arow = wm * 16 + l16;
#pragma unroll
        for (int j = 0; j < 4; ++j) fa.u[j]     = As[arow * 16 + hl * 4 + j];
#pragma unroll
        for (int j = 0; j < 4; ++j) fa.u[4 + j] = As[arow * 16 + 8 + hl * 4 + j];

#pragma unroll
        for (int ni = 0; ni < 8; ++ni) {
            FragU fb;
            const int bcol = ni * 16 + l16;
#pragma unroll
            for (int j = 0; j < 8; ++j) fb.u[j] = Bs[bcol * 16 + (j >> 2) * 8 + hl * 4 + (j & 3)];
            acc[ni] = __builtin_amdgcn_wmma_f32_16x16x32_bf16(
                false, fa.b, false, fb.b, (short)0, acc[ni], false, false);
        }
        __syncthreads();
    }

#pragma unroll
    for (int ni = 0; ni < 8; ++ni) {
        const int d = nb * 128 + ni * 16 + l16;
        const float bias = b2[e * DIM + d];
#pragma unroll
        for (int j = 0; j < 8; ++j) {
            int rl = wm * 16 + hl * 8 + j;
            if (mb * MBLK + rl < cnt_e) {
                int t = toks[rl];
                out[(size_t)t * DIM + d] = pg[t] * (acc[ni][j] + bias);
            }
        }
    }
}

// ---------------------------------------------------------------- launch
extern "C" void kernel_launch(void* const* d_in, const int* in_sizes, int n_in,
                              void* d_out, int out_size, void* d_ws, size_t ws_size,
                              hipStream_t stream) {
    const float* X  = (const float*)d_in[0];
    // d_in[1] input_ids (int64), d_in[2] attention_mask: unused by the math
    const float* Wg = (const float*)d_in[3];
    const float* W1 = (const float*)d_in[4];
    const float* b1 = (const float*)d_in[5];
    const float* W2 = (const float*)d_in[6];
    const float* b2 = (const float*)d_in[7];
    float* out = (float*)d_out;

    char* ws = (char*)d_ws;
    const size_t h_bytes = (size_t)TOKENS * FFN * 2;   // 32 MB bf16 hidden
    unsigned short* h = (unsigned short*)ws;
    int*   cnt    = (int*)  (ws + h_bytes);
    float* psum   = (float*)(ws + h_bytes + 64);
    int*   offs   = (int*)  (ws + h_bytes + 128);
    int*   cursor = (int*)  (ws + h_bytes + 192);
    int*   eid    = (int*)  (ws + h_bytes + 256);
    float* pg     = (float*)(ws + h_bytes + 256 + 4 * (size_t)TOKENS);
    int*   tl     = (int*)  (ws + h_bytes + 256 + 8 * (size_t)TOKENS);

    moe_init<<<1, 64, 0, stream>>>(cnt, psum, cursor);
    moe_gate<<<TOKENS / 8, 256, 0, stream>>>(X, Wg, eid, pg, cnt, psum);
    moe_scan<<<1, 32, 0, stream>>>(cnt, psum, offs, out);
    moe_scatter<<<TOKENS / 256, 256, 0, stream>>>(eid, cursor, offs, tl);
    moe_ffn1<<<dim3(NEXP * MAXMB, FFN / 128), 256, 0, stream>>>(X, W1, b1, cnt, offs, tl, h);
    moe_ffn2<<<dim3(NEXP * MAXMB, DIM / 128), 256, 0, stream>>>((const uint32_t*)h, W2, b2,
                                                                cnt, offs, tl, pg, out);
}
